// RPN1D_6219112644764
// MI455X (gfx1250) — matrix operand order repaired and checked by reading, not compiled, not run on hardware
//
#include <hip/hip_runtime.h>
#include <hip/hip_bf16.h>

typedef _Float16 v16h __attribute__((ext_vector_type(16)));
typedef _Float16 v8h  __attribute__((ext_vector_type(8)));
typedef float    v8f  __attribute__((ext_vector_type(8)));
typedef int      v4i  __attribute__((vector_size(16)));   // matches builtin param type

#define BATCH 8
#define CCH   512
#define LFEAT 8192
#define NA    7
#define KTOT  1536            // C*3
#define NLA   (LFEAT * NA)    // 57344
#define PRE_K 600
#define POST_K 100

#if __has_builtin(__builtin_amdgcn_global_load_async_to_lds_b128) && \
    __has_builtin(__builtin_amdgcn_s_wait_asynccnt)
#define USE_ASYNC_LDS 1
#else
#define USE_ASYNC_LDS 0
#endif

#define AS1 __attribute__((address_space(1)))
#define AS3 __attribute__((address_space(3)))

__constant__ float c_lens[NA] = {1.f, 2.f, 3.f, 4.f, 5.f, 7.f, 9.f};

// ---------- fragment helpers (CDNA5 WMMA f16 16x16x32 layouts) ----------
__device__ __forceinline__ v16h frag_a(const _Float16* row, int koff) {
    // A 16x32: lane holds 16 contiguous K = [koff..koff+15]
    v16h r;
    ((v8h*)&r)[0] = *(const v8h*)(row + koff);
    ((v8h*)&r)[1] = *(const v8h*)(row + koff + 8);
    return r;
}
__device__ __forceinline__ v16h frag_b(const _Float16* p) {
    // B 32x16: lane = column, 16 contiguous K values
    v16h r;
    ((v8h*)&r)[0] = *(const v8h*)(p);
    ((v8h*)&r)[1] = *(const v8h*)(p + 8);
    return r;
}

// ---------- prep kernels ----------
__global__ void cvt_w_kernel(const float* __restrict__ w, _Float16* __restrict__ o, int n) {
    int gi = blockIdx.x * 256 + threadIdx.x;
    if (gi < n) o[gi] = (_Float16)w[gi];
}

__global__ void prep_wh_kernel(const float* __restrict__ objw, const float* __restrict__ regw,
                               _Float16* __restrict__ o) {
    int gi = blockIdx.x * 256 + threadIdx.x;
    if (gi >= 32 * CCH) return;
    int n = gi / CCH, c = gi % CCH;
    float v = 0.f;
    if (n < 7)        v = objw[n * CCH + c];
    else if (n < 21)  v = regw[(n - 7) * CCH + c];
    o[gi] = (_Float16)v;
}

// ---------- fused conv (implicit GEMM, WMMA) + obj/reg heads (WMMA) ----------
// grid = (LFEAT/64, BATCH), block = 256 (8 waves, wave32)
// Each block: full 512 out-channels x 64 positions; each wave 64co x 64l (16 WMMA tiles).
__global__ __launch_bounds__(256)
void conv_head_kernel(const float* __restrict__ feat,
                      const _Float16* __restrict__ wf16,   // [512][1536] (co, ci*3+k)
                      const float* __restrict__ conv_b,
                      const _Float16* __restrict__ whf16,  // [32][512] head weights, padded
                      const float* __restrict__ obj_b,
                      const float* __restrict__ reg_b,
                      float* __restrict__ obj_out,         // [B][LF*A]
                      float* __restrict__ reg_out)         // [B][LF*A*2]
{
    const int tid = threadIdx.x;
    const int l0  = blockIdx.x * 64;
    const int b   = blockIdx.y;

    const int wid    = tid >> 5;          // wave32
    const int lane   = tid & 31;
    const int nlo    = lane & 15;
    const int hi     = (lane >> 4) & 1;   // lane group
    const int koffA  = hi ? 16 : 0;       // A-frag K offset
    const int koffB  = hi ? 16 : 0;       // B-frag K offset
    const int rowoff = hi ? 8 : 0;        // C/D row offset

    // LDS: W tile [512][40] (40960 B) + X tile [64][40] (5120 B)
    // head passes reuse the W region as H tile [32][520] (33280 B)
    __shared__ __align__(16) _Float16 smem[CCH * 40 + 64 * 40];
    _Float16* sW = smem;
    _Float16* sX = smem + CCH * 40;
    _Float16* sH = smem;

    const int co_base = wid * 64;
    v8f acc[4][4] = {};

    const int xn  = tid & 63;             // stage-X assignment: 64 l x 32 kt = 256 thr x 8
    const int xk0 = (tid >> 6) * 8;

    for (int kb = 0; kb < KTOT; kb += 32) {
        __syncthreads();  // previous iteration's LDS reads complete
        // ---- stage W tile: 512x32 halfs ----
#if USE_ASYNC_LDS
        #pragma unroll
        for (int i = 0; i < 8; ++i) {
            int q  = tid + 256 * i;
            int co = q >> 2, h = q & 3;               // 8-half (16B) chunks
            __builtin_amdgcn_global_load_async_to_lds_b128(
                (AS1 v4i*)(wf16 + co * KTOT + kb + h * 8),
                (AS3 v4i*)(sW + co * 40 + h * 8), 0, 0);
        }
#else
        #pragma unroll
        for (int i = 0; i < 4; ++i) {
            int q  = tid + 256 * i;
            int co = q >> 1, h = q & 1;
            *(v16h*)(sW + co * 40 + h * 16) =
                *(const v16h*)(wf16 + co * KTOT + kb + h * 16);
        }
#endif
        // ---- stage X tile (im2row, f32->f16): sX[l][kt] ----
        #pragma unroll
        for (int j = 0; j < 8; ++j) {
            int kt  = xk0 + j;
            int ktg = kb + kt;
            int ci  = ktg / 3;
            int kk  = ktg - ci * 3;
            int l   = l0 + xn + kk - 1;
            float v = ((unsigned)l < (unsigned)LFEAT)
                        ? feat[((size_t)b * CCH + ci) * LFEAT + l] : 0.f;
            sX[xn * 40 + kt] = (_Float16)v;
        }
        if (kb + 32 < KTOT)
            __builtin_prefetch(&feat[((size_t)b * CCH + (unsigned)(kb + 32) / 3) * LFEAT + l0 + xn], 0, 1);
#if USE_ASYNC_LDS
        __builtin_amdgcn_s_wait_asynccnt(0);
#endif
        __syncthreads();

        v16h bf[4];
        #pragma unroll
        for (int nt = 0; nt < 4; ++nt)
            bf[nt] = frag_b(sX + (nt * 16 + nlo) * 40 + koffB);
        #pragma unroll
        for (int t = 0; t < 4; ++t) {
            v16h af = frag_a(sW + (co_base + t * 16 + nlo) * 40, koffA);
            #pragma unroll
            for (int nt = 0; nt < 4; ++nt)
                acc[t][nt] = __builtin_amdgcn_wmma_f32_16x16x32_f16(
                    false, af, false, bf[nt], (short)0, acc[t][nt], false, false);
        }
    }

    // ---- fused heads, two passes of 32 positions over reused LDS ----
    for (int hp = 0; hp < 2; ++hp) {
        __syncthreads();   // sW/sX (or previous sH) reads complete
        // bias + relu, store transposed slice sH[l][c], l in [hp*32, hp*32+32)
        #pragma unroll
        for (int t = 0; t < 4; ++t) {
            #pragma unroll
            for (int p = 0; p < 2; ++p) {
                int nt = hp * 2 + p;
                #pragma unroll
                for (int r = 0; r < 8; ++r) {
                    int co = co_base + t * 16 + r + rowoff;
                    int ll = p * 16 + nlo;
                    float v = acc[t][nt][r] + conv_b[co];
                    v = fmaxf(v, 0.f);
                    sH[ll * 520 + co] = (_Float16)v;
                }
            }
        }
        __syncthreads();

        // heads: D(32 l x 32 n) = X(32 x 512) * Wh(512 x 32); waves 0..3, one tile each
        if (wid < 4) {
            const int mt  = wid >> 1;
            const int nt2 = wid & 1;
            v8f hacc = {};
            #pragma unroll
            for (int kb2 = 0; kb2 < CCH; kb2 += 32) {
                v16h af = frag_a(sH + (mt * 16 + nlo) * 520 + kb2, koffA);
                v16h bfh = frag_b(whf16 + (nt2 * 16 + nlo) * CCH + kb2 + koffB);
                hacc = __builtin_amdgcn_wmma_f32_16x16x32_f16(
                    false, af, false, bfh, (short)0, hacc, false, false);
            }
            #pragma unroll
            for (int r = 0; r < 8; ++r) {
                int lloc = mt * 16 + r + rowoff;
                int n    = nt2 * 16 + nlo;
                int gl   = l0 + hp * 32 + lloc;
                float v  = hacc[r];
                if (n < 7) {
                    obj_out[(size_t)b * NLA + gl * NA + n] = v + obj_b[n];
                } else if (n < 21) {
                    int j = n - 7;
                    reg_out[(size_t)b * NLA * 2 + gl * 14 + j] = v + reg_b[j];
                }
            }
        }
    }
}

// ---------- anchors ----------
__global__ void anchors_kernel(float* __restrict__ anch) {
    int gi = blockIdx.x * 256 + threadIdx.x;
    if (gi >= NLA) return;
    int l = gi / NA, a = gi % NA;
    float cen = (float)l + 0.5f, hl = 0.5f * c_lens[a];
    anch[gi * 2]     = cen - hl;
    anch[gi * 2 + 1] = cen + hl;
}

// ---------- decode boxes + sigmoid scores ----------
__global__ void decode_kernel(const float* __restrict__ obj, const float* __restrict__ reg,
                              float* __restrict__ boxes, float* __restrict__ scores) {
    int gi = blockIdx.x * 256 + threadIdx.x;
    if (gi >= BATCH * NLA) return;
    int i = gi % NLA;
    int l = i / NA, a = i % NA;
    float a_c = (float)l + 0.5f;
    float a_w = fmaxf(c_lens[a], 1e-6f);
    float t_c = reg[(size_t)gi * 2];
    float t_w = fminf(fmaxf(reg[(size_t)gi * 2 + 1], -10.f), 10.f);
    float c = a_c + t_c * a_w;
    float w = a_w * expf(t_w);
    const float Lff = (float)LFEAT;
    float s = fminf(fmaxf(c - 0.5f * w, 0.f), Lff);
    float e = fminf(fmaxf(c + 0.5f * w, 0.f), Lff);
    e = fminf(fmaxf(e, s + 1e-3f), Lff);
    s = fmaxf(fminf(s, e - 1e-3f), 0.f);
    boxes[(size_t)gi * 2]     = s;
    boxes[(size_t)gi * 2 + 1] = e;
    scores[gi] = 1.f / (1.f + expf(-obj[gi]));
}

// ---------- per-batch: top-600 -> NMS -> top-100 ----------
__global__ __launch_bounds__(1024)
void propose_kernel(float* __restrict__ scores, const float* __restrict__ boxes,
                    float* __restrict__ props, float* __restrict__ pscores,
                    float* __restrict__ pmask) {
    const int b   = blockIdx.x;
    const int tid = threadIdx.x;
    float* sc = scores + (size_t)b * NLA;
    const float* bg = boxes + (size_t)b * NLA * 2;

    __shared__ float red_v[1024];
    __shared__ int   red_i[1024];
    __shared__ float ts[PRE_K];
    __shared__ int   ti[PRE_K];
    __shared__ float bxs[PRE_K * 2];
    __shared__ int   keep[PRE_K];
    __shared__ float sk[PRE_K];
    __shared__ int   s_sup;

    // phase A: iterative top-600 (stable: ties -> lower index)
    for (int it = 0; it < PRE_K; ++it) {
        float lv = -3e38f; int li = 0x7fffffff;
        #pragma unroll 4
        for (int k = 0; k < NLA / 1024; ++k) {
            int idx = tid + k * 1024;
            float v = sc[idx];
            if (v > lv || (v == lv && idx < li)) { lv = v; li = idx; }
        }
        red_v[tid] = lv; red_i[tid] = li;
        __syncthreads();
        for (int s = 512; s > 0; s >>= 1) {
            if (tid < s) {
                float ov = red_v[tid + s]; int oi = red_i[tid + s];
                if (ov > red_v[tid] || (ov == red_v[tid] && oi < red_i[tid])) {
                    red_v[tid] = ov; red_i[tid] = oi;
                }
            }
            __syncthreads();
        }
        if (tid == 0) {
            ti[it] = red_i[0]; ts[it] = red_v[0];
            sc[red_i[0]] = -3e38f;
        }
        __syncthreads();
    }

    // phase B: gather boxes, sequential NMS
    if (tid < PRE_K) {
        int j = ti[tid];
        bxs[tid * 2]     = bg[(size_t)j * 2];
        bxs[tid * 2 + 1] = bg[(size_t)j * 2 + 1];
        keep[tid] = (ts[tid] >= 0.1f) ? 1 : 0;
    }
    __syncthreads();
    for (int i = 1; i < PRE_K; ++i) {
        if (tid == 0) s_sup = 0;
        __syncthreads();
        if (tid < i && keep[tid]) {
            float si = bxs[i * 2],   ei = bxs[i * 2 + 1];
            float sj = bxs[tid * 2], ej = bxs[tid * 2 + 1];
            float inter = fmaxf(fminf(ei, ej) - fmaxf(si, sj), 0.f);
            float uni   = (ei - si) + (ej - sj) - inter;
            if (inter / fmaxf(uni, 1e-6f) > 0.5f) s_sup = 1;
        }
        __syncthreads();
        if (tid == 0 && s_sup) keep[i] = 0;
        __syncthreads();
    }

    // phase C: top-100 of kept
    if (tid < PRE_K) sk[tid] = keep[tid] ? ts[tid] : -1.f;
    __syncthreads();
    for (int it = 0; it < POST_K; ++it) {
        red_v[tid] = (tid < PRE_K) ? sk[tid] : -3e38f;
        red_i[tid] = tid;
        __syncthreads();
        for (int s = 512; s > 0; s >>= 1) {
            if (tid < s) {
                float ov = red_v[tid + s]; int oi = red_i[tid + s];
                if (ov > red_v[tid] || (ov == red_v[tid] && oi < red_i[tid])) {
                    red_v[tid] = ov; red_i[tid] = oi;
                }
            }
            __syncthreads();
        }
        if (tid == 0) {
            int j = red_i[0]; float s = red_v[0];
            int m = s > 0.f;
            int o = b * POST_K + it;
            props[o * 2]     = m ? bxs[j * 2]     : 0.f;
            props[o * 2 + 1] = m ? bxs[j * 2 + 1] : 0.f;
            pscores[o] = m ? s : 0.f;
            pmask[o]   = m ? 1.f : 0.f;
            sk[j] = -3e38f;
        }
        __syncthreads();
    }
}

extern "C" void kernel_launch(void* const* d_in, const int* in_sizes, int n_in,
                              void* d_out, int out_size, void* d_ws, size_t ws_size,
                              hipStream_t stream) {
    const float* feat   = (const float*)d_in[0];
    const float* conv_w = (const float*)d_in[1];
    const float* conv_b = (const float*)d_in[2];
    const float* obj_w  = (const float*)d_in[3];
    const float* obj_b  = (const float*)d_in[4];
    const float* reg_w  = (const float*)d_in[5];
    const float* reg_b  = (const float*)d_in[6];

    // workspace layout (16B aligned)
    char* ws = (char*)d_ws;
    _Float16* wf16  = (_Float16*)ws;                   // 512*1536 halfs = 1572864 B
    _Float16* whf16 = (_Float16*)(ws + 1572864);       // 32*512 halfs   = 32768 B
    float*    boxes = (float*)(ws + 1605632);          // 8*57344*2 f32  = 3670016 B
    float*    sco   = (float*)(ws + 5275648);          // 8*57344 f32    = 1835008 B

    // output layout: obj | reg | anchors | props | pscores | pmask
    float* o       = (float*)d_out;
    float* obj_out = o;
    float* reg_out = o + (size_t)BATCH * NLA;                 // 458752
    float* anch    = reg_out + (size_t)BATCH * NLA * 2;       // +917504
    float* props   = anch + (size_t)NLA * 2;                  // +114688
    float* pscores = props + BATCH * POST_K * 2;              // +1600
    float* pmask   = pscores + BATCH * POST_K;                // +800

    cvt_w_kernel<<<(CCH * KTOT + 255) / 256, 256, 0, stream>>>(conv_w, wf16, CCH * KTOT);
    prep_wh_kernel<<<(32 * CCH + 255) / 256, 256, 0, stream>>>(obj_w, reg_w, whf16);
    conv_head_kernel<<<dim3(LFEAT / 64, BATCH), 256, 0, stream>>>(
        feat, wf16, conv_b, whf16, obj_b, reg_b, obj_out, reg_out);
    anchors_kernel<<<(NLA + 255) / 256, 256, 0, stream>>>(anch);
    decode_kernel<<<(BATCH * NLA + 255) / 256, 256, 0, stream>>>(obj_out, reg_out, boxes, sco);
    propose_kernel<<<BATCH, 1024, 0, stream>>>(sco, boxes, props, pscores, pmask);
}
